// DiffMultiHeadedAttention_31258771980658
// MI455X (gfx1250) — compile-verified, matrix operand order checked by reading, not run on hardware
//
#include <hip/hip_runtime.h>

typedef __attribute__((ext_vector_type(16))) _Float16 v16h;
typedef __attribute__((ext_vector_type(8)))  float    v8f;

#define B_   2
#define T_   2048
#define C_   1024
#define H_   8
#define DH_  128
#define DHH_ 64
#define H2_  16
#define LAMBDA_INIT_F 0.7836057665f

// ---- Async global->LDS staging (CDNA5, ASYNCcnt) via inline asm -----------
// ISA 15.18.3: GLOBAL_LOAD_ASYNC_TO_LDS_B128 (opcode 98), SOPP S_WAIT_ASYNCCNT.
// VDST VGPR holds the LDS byte address, VADDR the 64-bit global address.
#define USE_ASYNC_LDS 1

#if USE_ASYNC_LDS
__device__ inline void async_ld_b128(unsigned ldsOff, const void* g) {
  asm volatile("global_load_async_to_lds_b128 %0, %1, off"
               :
               : "v"(ldsOff), "v"(g)
               : "memory");
}
__device__ inline void wait_async0() {
  asm volatile("s_wait_asynccnt 0x0" ::: "memory");
}
#else
__device__ inline void wait_async0() {}
#endif

// ---------------------------------------------------------------------------
// WMMA fragment gathers (ISA 7.12.2, wave32). For lane L, half j:
//   v = j>>1; k = (v>>2)*16 + (L>=16 ? 8 : 0) + (v&3)*2 + (j&1)
// Pattern makes each fragment a pair of contiguous 16B runs in LDS, which the
// compiler lowers to ds_load_b128 (verified in round-1/2 asm).
// ---------------------------------------------------------------------------
__device__ inline void load_frag_mk(const _Float16* base, int ld, v16h& f,
                                    int lane, int koff) {
  const int m  = lane & 15;
  const int hi = (lane >> 4) & 1;
#pragma unroll
  for (int j = 0; j < 16; ++j) {
    const int v = j >> 1;
    const int k = ((v >> 2) << 4) + (hi << 3) + ((v & 3) << 1) + (j & 1) + koff;
    f[j] = base[m * ld + k];
  }
}

__device__ inline void load_frag_km(const _Float16* base, int ld, v16h& f,
                                    int lane, int koff) {
  const int n  = lane & 15;
  const int hi = (lane >> 4) & 1;
#pragma unroll
  for (int j = 0; j < 16; ++j) {
    const int v = j >> 1;
    const int k = ((v >> 2) << 4) + (hi << 3) + ((v & 3) << 1) + (j & 1) + koff;
    f[j] = base[k * ld + n];
  }
}

__device__ inline v8f wmma16(const v16h& a, const v16h& b, const v8f& c) {
  return __builtin_amdgcn_wmma_f32_16x16x32_f16(false, a, false, b,
                                                (short)0, c, false, false);
}

// ---------------------------------------------------------------------------
// Stage one K (64x64 f16) + V (64x128 f16) tile into padded LDS buffers.
// Async path: per-lane 16B chunks, global -> LDS DMA, no VGPR round-trip.
// ---------------------------------------------------------------------------
__device__ inline void stage_kv(const char* Kb, const char* Vb, int kt0,
                                _Float16 (*Ksb)[DHH_ + 8],
                                _Float16 (*Vsb)[DH_ + 8], int tid) {
#if USE_ASYNC_LDS
#pragma unroll
  for (int c = 0; c < 4; ++c) {               // K: 512 chunks of 16B
    const int e = tid + (c << 7);
    const int n = e >> 3, cc = e & 7;
    async_ld_b128((unsigned)(size_t)&Ksb[n][cc << 3],
                  Kb + (size_t)(kt0 + n) * (C_ * 2) + ((size_t)cc << 4));
  }
#pragma unroll
  for (int c = 0; c < 8; ++c) {               // V: 1024 chunks of 16B
    const int e = tid + (c << 7);
    const int n = e >> 4, cc = e & 15;
    async_ld_b128((unsigned)(size_t)&Vsb[n][cc << 3],
                  Vb + (size_t)(kt0 + n) * (C_ * 2) + ((size_t)cc << 4));
  }
#else
  const _Float16* Kh = (const _Float16*)Kb;
  const _Float16* Vh = (const _Float16*)Vb;
  for (int e = tid; e < 64 * DHH_; e += 128) {
    const int n = e >> 6, d = e & 63;
    Ksb[n][d] = Kh[(size_t)(kt0 + n) * C_ + d];
  }
  for (int e = tid; e < 64 * DH_; e += 128) {
    const int n = e >> 7, d = e & 127;
    Vsb[n][d] = Vh[(size_t)(kt0 + n) * C_ + d];
  }
#endif
}

// ---------------------------------------------------------------------------
// Tiled GEMM: C[M,N] = A[M,K] @ B[K,N], f16 WMMA, f32 accumulate.
// Block 256 threads (8 waves), tile 128x64, K-step 32.
// Wave (wr=wid>>1, wc=wid&1) owns a 32x32 C tile: 2 A-frags x 2 B-frags
// -> 4 WMMA per K-step with full outer-product fragment reuse.
// ---------------------------------------------------------------------------
template <typename AT, typename OT>
__global__ __launch_bounds__(256) void gemm_wmma_k(
    const AT* __restrict__ A, const float* __restrict__ Bw,
    OT* __restrict__ Cmat, int M, int N, int K) {
  __shared__ _Float16 As[128][40];
  __shared__ _Float16 Bs[32][72];

  const int tid  = threadIdx.x;
  const int lane = tid & 31;
  const int wid  = tid >> 5;
  const int nBN  = N >> 6;
  const int bm   = blockIdx.x / nBN;
  const int bn   = blockIdx.x % nBN;
  const int m0   = bm << 7;
  const int n0   = bn << 6;
  const int wr   = wid >> 1;
  const int wc   = wid & 1;

  v8f acc00 = {}, acc01 = {}, acc10 = {}, acc11 = {};

  for (int k0 = 0; k0 < K; k0 += 32) {
#pragma unroll
    for (int i = 0; i < 16; ++i) {                 // A tile 128x32
      const int e = tid + (i << 8);
      const int r = e >> 5, cc = e & 31;
      As[r][cc] = (_Float16)(float)A[(size_t)(m0 + r) * K + k0 + cc];
    }
#pragma unroll
    for (int i = 0; i < 8; ++i) {                  // B tile 32x64
      const int e = tid + (i << 8);
      const int r = e >> 6, cc = e & 63;
      Bs[r][cc] = (_Float16)Bw[(size_t)(k0 + r) * N + n0 + cc];
    }
    if (k0 + 32 < K) {                             // L2 prefetch of next tiles
      __builtin_prefetch(&A[(size_t)(m0 + (tid & 127)) * K + k0 + 32], 0, 1);
      __builtin_prefetch(&Bw[(size_t)(k0 + 32 + (tid >> 3)) * N + n0 +
                             ((tid & 7) << 3)], 0, 1);
    }
    __syncthreads();

    v16h a0, a1, b0, b1;
    load_frag_mk(&As[(wr << 5)][0], 40, a0, lane, 0);
    load_frag_mk(&As[(wr << 5) + 16][0], 40, a1, lane, 0);
    load_frag_km(&Bs[0][(wc << 5)], 72, b0, lane, 0);
    load_frag_km(&Bs[0][(wc << 5) + 16], 72, b1, lane, 0);
    acc00 = wmma16(a0, b0, acc00);
    acc01 = wmma16(a0, b1, acc01);
    acc10 = wmma16(a1, b0, acc10);
    acc11 = wmma16(a1, b1, acc11);
    __syncthreads();
  }

  const int mlane = lane & 15;
  const int hi    = lane >> 4;
#pragma unroll
  for (int i = 0; i < 8; ++i) {
    const int r0 = m0 + (wr << 5) + i + (hi << 3);
    const int r1 = r0 + 16;
    const int cA = n0 + (wc << 5) + mlane;
    Cmat[(size_t)r0 * N + cA]      = static_cast<OT>(acc00[i]);
    Cmat[(size_t)r0 * N + cA + 16] = static_cast<OT>(acc01[i]);
    Cmat[(size_t)r1 * N + cA]      = static_cast<OT>(acc10[i]);
    Cmat[(size_t)r1 * N + cA + 16] = static_cast<OT>(acc11[i]);
  }
}

// ---------------------------------------------------------------------------
// lambda = exp(sum(lq1*lk1)) - exp(sum(lq2*lk2)) + LAMBDA_INIT   (one wave)
// ---------------------------------------------------------------------------
__global__ void lambda_kernel(const float* __restrict__ lq1,
                              const float* __restrict__ lq2,
                              const float* __restrict__ lk1,
                              const float* __restrict__ lk2,
                              float* __restrict__ lamOut) {
  const int lane = threadIdx.x;
  float s1 = lq1[lane] * lk1[lane] + lq1[lane + 32] * lk1[lane + 32];
  float s2 = lq2[lane] * lk2[lane] + lq2[lane + 32] * lk2[lane + 32];
#pragma unroll
  for (int off = 1; off < 32; off <<= 1) {
    s1 += __shfl_xor(s1, off, 32);
    s2 += __shfl_xor(s2, off, 32);
  }
  if (lane == 0) lamOut[0] = __expf(s1) - __expf(s2) + LAMBDA_INIT_F;
}

// ---------------------------------------------------------------------------
// Flash attention per half-head. Block = 128 threads (4 waves) handles one
// (b, half-head, 64-query-row tile). Wave owns 16 query rows + online-softmax
// state (16-lane-replicated, matching WMMA C layout) + [16x128] f32 O accum.
// K/V tiles are DOUBLE-BUFFERED: async global->LDS loads for tile nt+1 are
// issued right after the top barrier, while WMMA/softmax runs on tile nt;
// the next iteration's "s_wait_asynccnt 0 ; barrier" releases them.
// Causal: nt <= qt, diagonal tiles element-masked.
// ---------------------------------------------------------------------------
__global__ __launch_bounds__(128) void flash_half_attn(
    const _Float16* __restrict__ Qh, const _Float16* __restrict__ Kh,
    const _Float16* __restrict__ Vh, float* __restrict__ Ohalf) {
  __shared__ _Float16 Ks[2][64][DHH_ + 8];  // [buf][key n][dh k]  row = 144 B
  __shared__ _Float16 Vs[2][64][DH_ + 8];   // [buf][key k][dh n]  row = 272 B
  __shared__ _Float16 Ps[4][16][64 + 8];    // per-wave P scratch [row][key]

  const int tid   = threadIdx.x;
  const int lane  = tid & 31;
  const int wid   = tid >> 5;
  const int mlane = lane & 15;
  const int hi    = lane >> 4;

  const int nQT = T_ / 64;
  const int qt  = blockIdx.x % nQT;
  const int hh  = (blockIdx.x / nQT) % H2_;
  const int b   = blockIdx.x / (nQT * H2_);
  const int h   = hh >> 1;
  const int trow0 = qt * 64 + wid * 16;

  const char* Kb = (const char*)(Kh + ((size_t)(b * T_)) * C_ + hh * DHH_);
  const char* Vb = (const char*)(Vh + ((size_t)(b * T_)) * C_ + h * DH_);

  // Q fragments for rows trow0..+15, dh 0..63 (two K=32 chunks)
  v16h q0, q1;
  {
    const _Float16* qb = Qh + ((size_t)(b * T_ + trow0)) * C_ + hh * DHH_;
#pragma unroll
    for (int j = 0; j < 16; ++j) {
      const int v = j >> 1;
      const int k = ((v >> 2) << 4) + (hi << 3) + ((v & 3) << 1) + (j & 1);
      q0[j] = qb[(size_t)mlane * C_ + k];
      q1[j] = qb[(size_t)mlane * C_ + 32 + k];
    }
  }

  float mrow[8], lrow[8];
  v8f O[8];
  const v8f vzero = {};
#pragma unroll
  for (int i = 0; i < 8; ++i) { mrow[i] = -1e30f; lrow[i] = 0.f; }
#pragma unroll
  for (int dt = 0; dt < 8; ++dt) O[dt] = vzero;

  stage_kv(Kb, Vb, 0, Ks[0], Vs[0], tid);     // prologue: tile 0 in flight

  for (int nt = 0; nt <= qt; ++nt) {
    wait_async0();          // this wave's chunks of tile nt have landed
    __syncthreads();        // everyone's chunks have landed

    if (nt < qt)            // overlap: issue tile nt+1 while computing nt
      stage_kv(Kb, Vb, (nt + 1) * 64, Ks[(nt + 1) & 1], Vs[(nt + 1) & 1], tid);

    const int buf = nt & 1;
    const int kt0 = nt * 64;

    // ---- scores S = (Q K^T) / 8 for 4 column tiles
    v8f S[4];
    const bool needMask = (kt0 + 63 > trow0);
#pragma unroll
    for (int ct = 0; ct < 4; ++ct) {
      v16h bk0, bk1;
      load_frag_mk(&Ks[buf][ct << 4][0], DHH_ + 8, bk0, lane, 0);
      load_frag_mk(&Ks[buf][ct << 4][0], DHH_ + 8, bk1, lane, 32);
      v8f s = vzero;
      s = wmma16(q0, bk0, s);
      s = wmma16(q1, bk1, s);
#pragma unroll
      for (int i = 0; i < 8; ++i) {
        float val = s[i] * 0.125f;
        if (needMask) {
          const int row = trow0 + i + (hi << 3);
          const int col = kt0 + (ct << 4) + mlane;
          if (col > row) val = -1e30f;
        }
        s[i] = val;
      }
      S[ct] = s;
    }

    // ---- online softmax update (rows live in 16-lane groups -> width=16)
#pragma unroll
    for (int i = 0; i < 8; ++i) {
      float tmax = fmaxf(fmaxf(S[0][i], S[1][i]), fmaxf(S[2][i], S[3][i]));
#pragma unroll
      for (int off = 1; off < 16; off <<= 1)
        tmax = fmaxf(tmax, __shfl_xor(tmax, off, 16));
      const float mnew = fmaxf(mrow[i], tmax);
      const float corr = __expf(mrow[i] - mnew);
      mrow[i] = mnew;
      lrow[i] *= corr;
#pragma unroll
      for (int dt = 0; dt < 8; ++dt) O[dt][i] *= corr;

      float rs = 0.f;
#pragma unroll
      for (int ct = 0; ct < 4; ++ct) {
        const float p = __expf(S[ct][i] - mnew);
        rs += p;
        Ps[wid][i + (hi << 3)][(ct << 4) + mlane] = (_Float16)p;
      }
#pragma unroll
      for (int off = 1; off < 16; off <<= 1) rs += __shfl_xor(rs, off, 16);
      lrow[i] += rs;
    }

    // ---- O += P @ V   (same-wave LDS RAW: DS ops are in-order per wave)
    v16h pa0, pa1;
    load_frag_mk(&Ps[wid][0][0], 64 + 8, pa0, lane, 0);
    load_frag_mk(&Ps[wid][0][0], 64 + 8, pa1, lane, 32);
#pragma unroll
    for (int dt = 0; dt < 8; ++dt) {
      v16h bv0, bv1;
      load_frag_km(&Vs[buf][0][dt << 4], DH_ + 8, bv0, lane, 0);
      load_frag_km(&Vs[buf][32][dt << 4], DH_ + 8, bv1, lane, 0);
      O[dt] = wmma16(pa0, bv0, O[dt]);
      O[dt] = wmma16(pa1, bv1, O[dt]);
    }
    // no trailing barrier: buffer `buf` is only overwritten by the staging
    // issued AFTER the next iteration's top barrier.
  }

  // ---- normalize and write O_half[b, hh, t, d]
#pragma unroll
  for (int dt = 0; dt < 8; ++dt) {
#pragma unroll
    for (int i = 0; i < 8; ++i) {
      const int row = trow0 + i + (hi << 3);
      const float o = O[dt][i] / lrow[i];
      Ohalf[(((size_t)(b * H2_ + hh) * T_ + row)) * DH_ + (dt << 4) + mlane] = o;
    }
  }
}

// ---------------------------------------------------------------------------
// out = O0 - lam*O1, per-head RMSNorm, * rms_weight * (1-LAMBDA_INIT), -> f16
// ---------------------------------------------------------------------------
__global__ __launch_bounds__(256) void combine_rms(
    const float* __restrict__ Ohalf, const float* __restrict__ lamP,
    const float* __restrict__ rmsW, _Float16* __restrict__ Y) {
  const int lane = threadIdx.x & 31;
  const int wid  = threadIdx.x >> 5;
  const size_t ridx = (size_t)blockIdx.x * 8 + wid;   // over B*H*T
  const int t = (int)(ridx % T_);
  const int h = (int)((ridx / T_) % H_);
  const int b = (int)(ridx / ((size_t)T_ * H_));

  const float lam = lamP[0];
  const float* o0 = Ohalf + (((size_t)(b * H2_ + 2 * h) * T_ + t)) * DH_;
  const float* o1 = Ohalf + (((size_t)(b * H2_ + 2 * h + 1) * T_ + t)) * DH_;

  float v[4];
  float ss = 0.f;
#pragma unroll
  for (int j = 0; j < 4; ++j) {
    const int d = lane + j * 32;
    v[j] = o0[d] - lam * o1[d];
    ss += v[j] * v[j];
  }
#pragma unroll
  for (int off = 1; off < 32; off <<= 1) ss += __shfl_xor(ss, off, 32);
  const float rms = rsqrtf(ss * (1.0f / DH_) + 1e-5f);
  const float scale = 1.0f - LAMBDA_INIT_F;
#pragma unroll
  for (int j = 0; j < 4; ++j) {
    const int d = lane + j * 32;
    Y[((size_t)(b * T_ + t)) * C_ + h * DH_ + d] =
        (_Float16)(v[j] * rms * rmsW[d] * scale);
  }
}

// ---------------------------------------------------------------------------
extern "C" void kernel_launch(void* const* d_in, const int* in_sizes, int n_in,
                              void* d_out, int out_size, void* d_ws,
                              size_t ws_size, hipStream_t stream) {
  const float* query = (const float*)d_in[0];
  const float* key_t = (const float*)d_in[1];
  const float* value = (const float*)d_in[2];
  const float* Wq    = (const float*)d_in[3];
  const float* Wk    = (const float*)d_in[4];
  const float* Wv    = (const float*)d_in[5];
  const float* Wo    = (const float*)d_in[6];
  const float* lq1   = (const float*)d_in[7];
  const float* lq2   = (const float*)d_in[8];
  const float* lk1   = (const float*)d_in[9];
  const float* lk2   = (const float*)d_in[10];
  const float* rmsW  = (const float*)d_in[11];

  char* ws = (char*)d_ws;
  _Float16* Qh   = (_Float16*)(ws);                       //  8 MB
  _Float16* Kh   = (_Float16*)(ws + ((size_t)8  << 20));  //  8 MB
  _Float16* Vh   = (_Float16*)(ws + ((size_t)16 << 20));  //  8 MB
  _Float16* Y    = (_Float16*)(ws + ((size_t)24 << 20));  //  8 MB
  float*    Ohlf = (float*)   (ws + ((size_t)32 << 20));  // 32 MB
  float*    lam  = (float*)   (ws + ((size_t)64 << 20));  //  4 B

  const int M = B_ * T_;                  // 4096
  const dim3 gG((M / 128) * (C_ / 64));   // 512 blocks
  const dim3 bG(256);

  gemm_wmma_k<float, _Float16><<<gG, bG, 0, stream>>>(query, Wq, Qh, M, C_, C_);
  gemm_wmma_k<float, _Float16><<<gG, bG, 0, stream>>>(key_t, Wk, Kh, M, C_, C_);
  gemm_wmma_k<float, _Float16><<<gG, bG, 0, stream>>>(value, Wv, Vh, M, C_, C_);

  lambda_kernel<<<1, 32, 0, stream>>>(lq1, lq2, lk1, lk2, lam);

  flash_half_attn<<<B_ * H2_ * (T_ / 64), 128, 0, stream>>>(Qh, Kh, Vh, Ohlf);

  combine_rms<<<(B_ * H_ * T_) / 8, 256, 0, stream>>>(Ohlf, lam, rmsW, Y);

  gemm_wmma_k<_Float16, float><<<gG, bG, 0, stream>>>(Y, Wo, (float*)d_out,
                                                      M, C_, C_);
}
